// GCNConv_22325240004644
// MI455X (gfx1250) — matrix operand order repaired and checked by reading, not compiled, not run on hardware
//
#include <hip/hip_runtime.h>

// GCN conv: out = D^-1/2 (A+I) D^-1/2 (x W^T + b)
// N=16384, IN_DIM=512, OUT_DIM=256, all fp32 in/out.
// out[i] = dis[i]*( sum_j A[i,j]*(dis[j]*h[j]) + dis[i]*h[i] ), dis = rsqrt(deg).
// Big 137-GFLOP aggregation runs on v_wmma_f32_16x16x32_f16 (f16 ops, f32 acc) so the
// kernel is HBM-bound on the 2x 1.07GB adjacency sweeps (~92us floor @ 23.3 TB/s).
// k_aggregate is double-buffered + software-pipelined; B tiles use async-to-LDS.

typedef __attribute__((ext_vector_type(16))) _Float16 v16h;
typedef __attribute__((ext_vector_type(8)))  _Float16 v8h;
typedef __attribute__((ext_vector_type(8)))  float    v8f;
typedef __attribute__((ext_vector_type(4)))  int      v4i;

#define NN      16384
#define IN_DIM  512
#define OUT_DIM 256
#define LDSW    40   // LDS row stride in halves (80B: 16B-aligned, bank-conflict-free)

#if defined(__has_builtin)
#if __has_builtin(__builtin_amdgcn_global_load_async_to_lds_b128)
#define HAVE_ASYNC_LDS 1
#endif
#endif

#ifdef HAVE_ASYNC_LDS
// Builtin prototype (from hipcc diagnostic): (v4i*, v4i*, imm offset, imm cpol);
// offset is added to BOTH the global and LDS addresses (ISA IOFFSET semantics).
#define ASYNC_CP16(gp, lp, off) \
  __builtin_amdgcn_global_load_async_to_lds_b128((v4i*)(gp), (v4i*)(lp), (off), 0)
static __device__ __forceinline__ void async_wait0() {
#if __has_builtin(__builtin_amdgcn_s_wait_asynccnt)
  __builtin_amdgcn_s_wait_asynccnt(0);
#else
  asm volatile("s_wait_asynccnt 0x0" ::: "memory");
#endif
}
#endif

static __device__ __forceinline__ v16h cat8(v8h lo, v8h hi) {
  return __builtin_shufflevector(lo, hi, 0,1,2,3,4,5,6,7,8,9,10,11,12,13,14,15);
}

// ---------------------------------------------------------------------------
// Kernel 1: dis[i] = rsqrt(1 + sum_j A[i,j]) ; one block per row, streams 1GB.
// ---------------------------------------------------------------------------
__global__ __launch_bounds__(256) void k_degree(const float* __restrict__ A,
                                                float* __restrict__ dis) {
  __shared__ float red[256];
  const int row = blockIdx.x;
  const int tid = threadIdx.x;
  const float4* rp = (const float4*)(A + (size_t)row * NN);
  float s = 0.0f;
#pragma unroll
  for (int i = 0; i < 16; ++i) {          // 16 * 256 * 4 = 16384 floats
    float4 v = rp[tid + i * 256];
    s += v.x + v.y + v.z + v.w;
  }
  red[tid] = s;
  __syncthreads();
  for (int off = 128; off > 0; off >>= 1) {
    if (tid < off) red[tid] += red[tid + off];
    __syncthreads();
  }
  if (tid == 0) dis[row] = rsqrtf(red[0] + 1.0f);
}

// ---------------------------------------------------------------------------
// Kernel 2: h = x @ W^T + b (f32 out) and h2t[c][i] = f16(dis[i]*h[i][c]).
// WMMA GEMM M=16384 N=256 K=512; 64-row tile per block, 8 waves. (Small: 4.3GF)
// ---------------------------------------------------------------------------
__global__ __launch_bounds__(256) void k_linear(const float* __restrict__ x,
                                                const float* __restrict__ W,
                                                const float* __restrict__ b,
                                                const float* __restrict__ dis,
                                                float* __restrict__ h,
                                                _Float16* __restrict__ h2t) {
  __shared__ _Float16 sA[64 * LDSW];        // 64 rows x 32 K
  __shared__ _Float16 sB[OUT_DIM * LDSW];   // 256 channels x 32 K
  const int tid   = threadIdx.x;
  const int lane  = tid & 31;
  const int wave  = tid >> 5;               // 0..7
  const int mtile = wave >> 1;              // 0..3 : which 16-row tile
  const int nhalf = wave & 1;               // 0..1 : which 128-col half
  const int rowBase = blockIdx.x * 64;
  const int a_row = tid >> 2;               // 0..63
  const int a_col = (tid & 3) * 8;          // 0,8,16,24

  v8f acc[8];
#pragma unroll
  for (int i = 0; i < 8; ++i)
#pragma unroll
    for (int j = 0; j < 8; ++j) acc[i][j] = 0.0f;

  for (int k0 = 0; k0 < IN_DIM; k0 += 32) {
    const float4* ap = (const float4*)(x + (size_t)(rowBase + a_row) * IN_DIM + k0 + a_col);
    float4 f0 = ap[0], f1 = ap[1];
    _Float16* da = &sA[a_row * LDSW + a_col];
    da[0] = (_Float16)f0.x; da[1] = (_Float16)f0.y; da[2] = (_Float16)f0.z; da[3] = (_Float16)f0.w;
    da[4] = (_Float16)f1.x; da[5] = (_Float16)f1.y; da[6] = (_Float16)f1.z; da[7] = (_Float16)f1.w;
    const float4* wp = (const float4*)(W + (size_t)tid * IN_DIM + k0);
    _Float16* db = &sB[tid * LDSW];
#pragma unroll
    for (int j = 0; j < 8; ++j) {
      float4 w4 = wp[j];
      db[j*4+0] = (_Float16)w4.x; db[j*4+1] = (_Float16)w4.y;
      db[j*4+2] = (_Float16)w4.z; db[j*4+3] = (_Float16)w4.w;
    }
    __syncthreads();
    const _Float16* afb = &sA[(mtile*16 + (lane & 15)) * LDSW + ((lane >> 4) * 8)];
    v16h afrag = cat8(*(const v8h*)afb, *(const v8h*)(afb + 16));
#pragma unroll
    for (int nt = 0; nt < 8; ++nt) {
      const int Cb = nhalf * 128 + nt * 16;
      const _Float16* bfb = &sB[(Cb + (lane & 15)) * LDSW + ((lane >> 4) * 8)];
      v16h bfrag = cat8(*(const v8h*)bfb, *(const v8h*)(bfb + 16));
      acc[nt] = __builtin_amdgcn_wmma_f32_16x16x32_f16(
          false, afrag, false, bfrag, (short)0, acc[nt], false, false);
    }
    __syncthreads();
  }
#pragma unroll
  for (int nt = 0; nt < 8; ++nt) {
    const int col = nhalf * 128 + nt * 16 + (lane & 15);
#pragma unroll
    for (int r = 0; r < 8; ++r) {
      const int row = rowBase + mtile * 16 + r + ((lane >> 4) * 8);
      float hv = acc[nt][r] + b[col];
      h[(size_t)row * OUT_DIM + col] = hv;
      h2t[(size_t)col * NN + row] = (_Float16)(hv * dis[row]);
    }
  }
}

// ---------------------------------------------------------------------------
// Kernel 3: out = dis[i] * ( A_f16 @ h2t + dis[i]*h[i] ). The big GEMM.
// 256 blocks x 64 rows. Double-buffered LDS, 1-deep pipeline: next A tile in
// registers + next B tile via async-to-LDS while WMMAs run on current tile.
// ---------------------------------------------------------------------------
__global__ __launch_bounds__(256) void k_aggregate(const float* __restrict__ A,
                                                   const float* __restrict__ h,
                                                   const _Float16* __restrict__ h2t,
                                                   const float* __restrict__ dis,
                                                   float* __restrict__ out) {
  __shared__ _Float16 sA[2][64 * LDSW];        // 2 x 5.0 KB
  __shared__ _Float16 sB[2][OUT_DIM * LDSW];   // 2 x 20.0 KB
  const int tid   = threadIdx.x;
  const int lane  = tid & 31;
  const int wave  = tid >> 5;
  const int mtile = wave >> 1;
  const int nhalf = wave & 1;
  const int rowBase = blockIdx.x * 64;
  const int a_row = tid >> 2;
  const int a_col = (tid & 3) * 8;

  const float*    aRow = A + (size_t)(rowBase + a_row) * NN + a_col;  // this thread's A slice
  const _Float16* bRow = h2t + (size_t)tid * NN;                      // this thread's channel

  v8f acc[8];
#pragma unroll
  for (int i = 0; i < 8; ++i)
#pragma unroll
    for (int j = 0; j < 8; ++j) acc[i][j] = 0.0f;

  // ---- prologue: stage tile k0 = 0 into buffer 0 ----
  {
    float4 f0 = ((const float4*)aRow)[0];
    float4 f1 = ((const float4*)aRow)[1];
    _Float16* da = &sA[0][a_row * LDSW + a_col];
    da[0] = (_Float16)f0.x; da[1] = (_Float16)f0.y; da[2] = (_Float16)f0.z; da[3] = (_Float16)f0.w;
    da[4] = (_Float16)f1.x; da[5] = (_Float16)f1.y; da[6] = (_Float16)f1.z; da[7] = (_Float16)f1.w;
#ifdef HAVE_ASYNC_LDS
    _Float16* lb = &sB[0][tid * LDSW];
    ASYNC_CP16(bRow, lb,  0);
    ASYNC_CP16(bRow, lb, 16);
    ASYNC_CP16(bRow, lb, 32);
    ASYNC_CP16(bRow, lb, 48);
    async_wait0();
#else
    uint4 b0 = *(const uint4*)bRow;
    uint4 b1 = *(const uint4*)(bRow + 8);
    *(uint4*)&sB[0][tid * LDSW]     = b0;
    *(uint4*)&sB[0][tid * LDSW + 8] = b1;
#endif
  }
  __syncthreads();

  int buf = 0;
  for (int k0 = 0; k0 < NN; k0 += 32) {
    const int kn = k0 + 32;
    const int nb = buf ^ 1;
    const bool more = (kn < NN);

    // ---- issue next tile's loads (overlap with compute below) ----
    float4 nf0 = {}, nf1 = {};
    if (more) {
      nf0 = ((const float4*)(aRow + kn))[0];
      nf1 = ((const float4*)(aRow + kn))[1];
      __builtin_prefetch(aRow + kn + 64, 0, 0);   // 2 tiles ahead -> global_prefetch_b8
#ifdef HAVE_ASYNC_LDS
      const _Float16* bp = bRow + kn;
      _Float16* lb = &sB[nb][tid * LDSW];
      ASYNC_CP16(bp, lb,  0);
      ASYNC_CP16(bp, lb, 16);
      ASYNC_CP16(bp, lb, 32);
      ASYNC_CP16(bp, lb, 48);
#endif
    }
#ifndef HAVE_ASYNC_LDS
    uint4 nb0 = {}, nb1 = {};
    if (more) {
      nb0 = *(const uint4*)(bRow + kn);
      nb1 = *(const uint4*)(bRow + kn + 8);
    }
#endif

    // ---- compute on buffer `buf` ----
    const _Float16* afb = &sA[buf][(mtile*16 + (lane & 15)) * LDSW + ((lane >> 4) * 8)];
    v16h afrag = cat8(*(const v8h*)afb, *(const v8h*)(afb + 16));
#pragma unroll
    for (int nt = 0; nt < 8; ++nt) {
      const int Cb = nhalf * 128 + nt * 16;
      const _Float16* bfb = &sB[buf][(Cb + (lane & 15)) * LDSW + ((lane >> 4) * 8)];
      v16h bfrag = cat8(*(const v8h*)bfb, *(const v8h*)(bfb + 16));
      acc[nt] = __builtin_amdgcn_wmma_f32_16x16x32_f16(
          false, afrag, false, bfrag, (short)0, acc[nt], false, false);
    }

    // ---- commit next tile into buffer `nb` ----
    if (more) {
      _Float16* da = &sA[nb][a_row * LDSW + a_col];
      da[0] = (_Float16)nf0.x; da[1] = (_Float16)nf0.y; da[2] = (_Float16)nf0.z; da[3] = (_Float16)nf0.w;
      da[4] = (_Float16)nf1.x; da[5] = (_Float16)nf1.y; da[6] = (_Float16)nf1.z; da[7] = (_Float16)nf1.w;
#ifndef HAVE_ASYNC_LDS
      *(uint4*)&sB[nb][tid * LDSW]     = nb0;
      *(uint4*)&sB[nb][tid * LDSW + 8] = nb1;
#endif
    }
#ifdef HAVE_ASYNC_LDS
    if (more) async_wait0();
#endif
    __syncthreads();
    buf = nb;
  }

  // ---- epilogue: out[i,c] = dis[i]*(acc + dis[i]*h[i,c]) (self-loop folded) ----
#pragma unroll
  for (int nt = 0; nt < 8; ++nt) {
    const int col = nhalf * 128 + nt * 16 + (lane & 15);
#pragma unroll
    for (int r = 0; r < 8; ++r) {
      const int row = rowBase + mtile * 16 + r + ((lane >> 4) * 8);
      const float d = dis[row];
      out[(size_t)row * OUT_DIM + col] =
          d * (acc[nt][r] + d * h[(size_t)row * OUT_DIM + col]);
    }
  }
}

// ---------------------------------------------------------------------------
extern "C" void kernel_launch(void* const* d_in, const int* in_sizes, int n_in,
                              void* d_out, int out_size, void* d_ws, size_t ws_size,
                              hipStream_t stream) {
  const float* x   = (const float*)d_in[0];   // [16384,512]
  const float* adj = (const float*)d_in[1];   // [16384,16384]
  const float* W   = (const float*)d_in[2];   // [256,512]
  const float* b   = (const float*)d_in[3];   // [256]
  float* out = (float*)d_out;                 // [16384,256]

  // workspace: dis (64KB) | h f32 (16MB) | h2t f16 (8MB)
  float*    dis = (float*)d_ws;
  float*    h   = dis + NN;
  _Float16* h2t = (_Float16*)(h + (size_t)NN * OUT_DIM);

  k_degree   <<<NN,      256, 0, stream>>>(adj, dis);
  k_linear   <<<NN / 64, 256, 0, stream>>>(x, W, b, dis, h, h2t);
  k_aggregate<<<NN / 64, 256, 0, stream>>>(adj, h, h2t, dis, out);
}
